// NaiveLSTM_8856222564931
// MI455X (gfx1250) — matrix operand order repaired and checked by reading, not compiled, not run on hardware
//
#include <hip/hip_runtime.h>
#include <cstddef>

// ---------------------------------------------------------------------------
// Types for CDNA5 WMMA (wave32): D(f32 16x16) = A(bf16 16x32) * B(bf16 32x16) + C
// ---------------------------------------------------------------------------
typedef float v8f __attribute__((ext_vector_type(8)));
typedef __bf16 v16bf __attribute__((ext_vector_type(16)));
typedef unsigned int v4u __attribute__((ext_vector_type(4)));
typedef unsigned int u32x4 __attribute__((ext_vector_type(4)));
typedef int i32x8 __attribute__((ext_vector_type(8)));
typedef int i32x4 __attribute__((ext_vector_type(4)));

union BFrag { v4u u[2]; v16bf v; };   // 32 bytes = 16 bf16 per lane

#define HSZ   512      // hidden size
#define ISZ   512      // input size
#define BSZ   32       // batch
#define SSZ   2048     // sequence length
#define G4    2048     // 4*H (i,f,g,o gate columns)

__device__ __forceinline__ unsigned short f2bf(float f) {
  union { float f; unsigned u; } v; v.f = f;
  unsigned u = v.u;
  unsigned r = 0x7FFFu + ((u >> 16) & 1u);   // round-to-nearest-even
  return (unsigned short)((u + r) >> 16);
}
__device__ __forceinline__ float bf2f(unsigned short s) {
  union { unsigned u; float f; } v; v.u = ((unsigned)s) << 16; return v.f;
}

// ---------------------------------------------------------------------------
// Prep kernels: pack gate weights [4H][K] to bf16, fuse biases, convert X.
// Gate order matches reference concat: i(ii), f(if), g(ig), o(io).
// ---------------------------------------------------------------------------
__global__ void lstm_pack_w(const float* __restrict__ g0, const float* __restrict__ g1,
                            const float* __restrict__ g2, const float* __restrict__ g3,
                            unsigned short* __restrict__ dst) {
  int idx = blockIdx.x * blockDim.x + threadIdx.x;   // over 2048*512
  int n = idx >> 9, k = idx & 511;
  int gate = n >> 9, row = n & 511;
  const float* src = gate == 0 ? g0 : gate == 1 ? g1 : gate == 2 ? g2 : g3;
  dst[idx] = f2bf(src[row * 512 + k]);
}

__global__ void lstm_pack_bias(const float* __restrict__ a0, const float* __restrict__ a1,
                               const float* __restrict__ b0, const float* __restrict__ b1,
                               const float* __restrict__ c0, const float* __restrict__ c1,
                               const float* __restrict__ d0, const float* __restrict__ d1,
                               float* __restrict__ dst) {
  int idx = blockIdx.x * blockDim.x + threadIdx.x;   // over 2048
  int gate = idx >> 9, row = idx & 511;
  float v;
  if (gate == 0)      v = a0[row] + a1[row];
  else if (gate == 1) v = b0[row] + b1[row];
  else if (gate == 2) v = c0[row] + c1[row];
  else                v = d0[row] + d1[row];
  dst[idx] = v;
}

__global__ void lstm_conv_x(const float* __restrict__ x, unsigned short* __restrict__ dst, int n) {
  int idx = blockIdx.x * blockDim.x + threadIdx.x;
  if (idx < n) dst[idx] = f2bf(x[idx]);
}

// ---------------------------------------------------------------------------
// Kernel 1: x_proj[s][b][n] = bias[n] + sum_k X[b][s][k] * Wx[n][k]
// Grid (2048, 2, 2), 256 threads = 8 waves. Each wave: 16 rows x 128 cols.
// Zero-init accumulators (first WMMA can take inline C=0); bias in epilogue.
// ---------------------------------------------------------------------------
__global__ __launch_bounds__(256) void lstm_xproj_gemm(
    const unsigned short* __restrict__ Xb, const unsigned short* __restrict__ Wx,
    const float* __restrict__ bias, unsigned short* __restrict__ xp) {
  const int s    = blockIdx.x;
  const int mt   = blockIdx.y;          // batch tile (rows 0-15 / 16-31)
  const int zh   = blockIdx.z;          // which 1024-column half
  const int lane = threadIdx.x & 31;
  const int w    = threadIdx.x >> 5;
  const int half = lane >> 4;
  const int ln   = lane & 15;
  const int nw   = zh * 1024 + w * 128; // wave's first output column

  // A layout (bf16 16x32): lanes 0-15 & 16-31 both hold M=ln,
  // K = k0+16h+[0..15]; X layout [B][S][I].
  const unsigned short* arow = Xb + ((size_t)(mt * 16 + ln) * SSZ + s) * ISZ + half * 16;

  // Hoisted B row bases: lane's column row in Wx, shifted by the lane-half K split.
  const unsigned short* wrow[8];
  #pragma unroll
  for (int t = 0; t < 8; ++t)
    wrow[t] = Wx + (size_t)(nw + t * 16 + ln) * ISZ + half * 16;

  v8f acc[8];
  #pragma unroll
  for (int t = 0; t < 8; ++t)
    #pragma unroll
    for (int r = 0; r < 8; ++r) acc[t][r] = 0.f;

  #pragma unroll
  for (int k0 = 0; k0 < ISZ; k0 += 32) {
    BFrag a;
    a.u[0] = *(const v4u*)(arow + k0);
    a.u[1] = *(const v4u*)(arow + k0 + 8);
    #pragma unroll
    for (int t = 0; t < 8; ++t) {
      BFrag b;   // B layout (bf16 32x16): lane holds N, K = k0+half*16+[0..15]
      b.u[0] = *(const v4u*)(wrow[t] + k0);
      b.u[1] = *(const v4u*)(wrow[t] + k0 + 8);
      acc[t] = __builtin_amdgcn_wmma_f32_16x16x32_bf16(
          false, a.v, false, b.v, (short)0, acc[t], false, false);
    }
  }

  // D element (r, lane): M = r + 8*half, N = ln.  Add fused bias here.
  #pragma unroll
  for (int t = 0; t < 8; ++t) {
    const float bv = bias[nw + t * 16 + ln];
    #pragma unroll
    for (int r = 0; r < 8; ++r) {
      int b = mt * 16 + r + half * 8;
      int n = nw + t * 16 + ln;
      xp[((size_t)s * BSZ + b) * G4 + n] = f2bf(acc[t][r] + bv);
    }
  }
}

// ---------------------------------------------------------------------------
// Kernel 2: sequential scan. One persistent 32-wave workgroup on one WGP.
//   LDS (288 KB of the WGP's 320 KB):
//     gact : activated gates f16 [32][2048]            128 KB
//     hbf  : h state bf16      [32][512]                32 KB
//     xpl  : TDM-staged xp slice bf16 [32][2048]       128 KB
//   c state lives in registers of the updating threads (16 per thread).
// Per step: 32 waves x 8 tiles x 16 K-steps = 4096 v_wmma_f32_16x16x32_bf16.
// The xp slice for step s+1 is fetched by the Tensor Data Mover
// (tensor_load_to_lds, TENSORcnt) and overlapped with GEMM + elementwise.
// ---------------------------------------------------------------------------
__device__ __forceinline__ void tdm_load_xp(unsigned lds_addr,
                                            const unsigned short* gsrc) {
  const unsigned long long ga = (unsigned long long)(const void*)gsrc;
  // D# group 0: count=1 | lds_addr | global_addr[56:0] | type=2
  u32x4 g0;
  g0[0] = 0x1u;                                   // count=1, user mode
  g0[1] = lds_addr;                               // LDS byte address
  g0[2] = (unsigned)(ga & 0xFFFFFFFFu);           // global_addr[31:0]
  g0[3] = (unsigned)((ga >> 32) & 0x01FFFFFFu) | 0x80000000u;  // [56:32] | type=2
  // D# group 1: 2D tensor, data_size=2B, dim0=2048 (stride 2048), dim1=32,
  //             tile = 2048 x 32 (full slice), no pad/iterate/multicast.
  i32x8 g1;
  g1[0] = 0x00010000;      // workgroup_mask=0, data_size=1 (2 bytes)
  g1[1] = 0x08000000;      // tensor_dim0[15:0]=2048 in bits [63:48]
  g1[2] = 0x00200000;      // tensor_dim0[31:16]=0, tensor_dim1[15:0]=32
  g1[3] = 0x08000000;      // tensor_dim1[31:16]=0, tile_dim0=2048
  g1[4] = 32;              // tile_dim1=32, tile_dim2=0
  g1[5] = 2048;            // tensor_dim0_stride[31:0]=2048
  g1[6] = 0;               // stride high bits / tensor_dim1_stride low = 0
  g1[7] = 0;
  i32x4 gz4 = {0, 0, 0, 0};               // groups 2/3 unused (<=2D tensor)
  i32x8 gz8 = {0, 0, 0, 0, 0, 0, 0, 0};   // extra group (6-arg toolchain form)
  __builtin_amdgcn_tensor_load_to_lds(g0, g1, gz4, gz4, gz8, 0);
}

__global__ __launch_bounds__(1024) void lstm_scan(
    const unsigned short* __restrict__ Wh, const unsigned short* __restrict__ xpb,
    float* __restrict__ out) {
  extern __shared__ char smem[];
  _Float16*       gact = (_Float16*)smem;                               // [32][2048]
  unsigned short* hbf  = (unsigned short*)(smem + BSZ * G4 * 2);        // [32][512]
  unsigned short* xpl  = (unsigned short*)(smem + BSZ * G4 * 2 + BSZ * HSZ * 2);

  const int tid  = threadIdx.x;
  const int lane = tid & 31;
  const int w    = tid >> 5;
  const int half = lane >> 4;
  const int ln   = lane & 15;
  const int mt   = w >> 4;          // waves 0-15: rows 0-15; 16-31: rows 16-31
  const int nb   = (w & 15) * 8;    // first of 8 column tiles for this wave

  // Generic LDS pointer: low 32 bits are the wave-relative LDS byte offset.
  const unsigned xpl_base = (unsigned)(size_t)(void*)xpl;

  float c[16];
  #pragma unroll
  for (int q = 0; q < 16; ++q) c[q] = 0.f;
  for (int i = tid; i < BSZ * HSZ; i += 1024) hbf[i] = 0;  // h0 = 0
  __syncthreads();

  // Step-invariant operand bases.
  const unsigned short* hrow = hbf + (size_t)(mt * 16 + ln) * HSZ + half * 16;
  const unsigned short* wrow[8];
  #pragma unroll
  for (int t = 0; t < 8; ++t)
    wrow[t] = Wh + (size_t)((nb + t) * 16 + ln) * HSZ + half * 16;

  // Preload xp slice for s = 0 via the Tensor Data Mover (one op, wave 0).
  if (w == 0) tdm_load_xp(xpl_base, xpb);

  for (int s = 0; s < SSZ; ++s) {
    // xp slice for this step must be fully landed in LDS.
    if (w == 0) __builtin_amdgcn_s_wait_tensorcnt(0);
    __syncthreads();   // broadcast TDM completion; also orders hbf update vs GEMM

    // ---- accumulator init: gates = xp[s] (bias already fused) --------------
    v8f acc[8];
    #pragma unroll
    for (int t = 0; t < 8; ++t) {
      const int n0 = (nb + t) * 16;
      #pragma unroll
      for (int r = 0; r < 8; ++r)
        acc[t][r] = bf2f(xpl[(size_t)(mt * 16 + r + half * 8) * G4 + n0 + ln]);
    }
    __syncthreads();   // all waves consumed xpl[s]; safe to overwrite

    // ---- kick off TDM staging of xp slice for step s+1 ---------------------
    if (s + 1 < SSZ && w == 0)
      tdm_load_xp(xpl_base, xpb + (size_t)(s + 1) * BSZ * G4);

    // ---- recurrent GEMM: gates += h @ Wh^T ---------------------------------
    #pragma unroll
    for (int k0 = 0; k0 < HSZ; k0 += 32) {
      BFrag a;   // A fragment from LDS-resident h (bf16)
      a.u[0] = *(const v4u*)(hrow + k0);
      a.u[1] = *(const v4u*)(hrow + k0 + 8);
      #pragma unroll
      for (int t = 0; t < 8; ++t) {
        BFrag b;
        b.u[0] = *(const v4u*)(wrow[t] + k0);
        b.u[1] = *(const v4u*)(wrow[t] + k0 + 8);
        acc[t] = __builtin_amdgcn_wmma_f32_16x16x32_bf16(
            false, a.v, false, b.v, (short)0, acc[t], false, false);
      }
    }

    // ---- activations, scatter to LDS gate buffer --------------------------
    #pragma unroll
    for (int t = 0; t < 8; ++t) {
      const int n0 = (nb + t) * 16;
      const int gate = n0 >> 9;     // 0=i,1=f,2=g,3=o (uniform per 16-col tile)
      #pragma unroll
      for (int r = 0; r < 8; ++r) {
        float v = acc[t][r];
        v = (gate == 2) ? tanhf(v) : 1.f / (1.f + __expf(-v));
        gact[(size_t)(mt * 16 + r + half * 8) * G4 + n0 + ln] = (_Float16)v;
      }
    }
    __syncthreads();

    // ---- elementwise state update; c in registers -------------------------
    float* outs = out + (size_t)s * (BSZ * HSZ);
    #pragma unroll
    for (int q = 0; q < 16; ++q) {
      const int id = q * 1024 + tid;      // coalesced over tid
      const int b = id >> 9, j = id & 511;
      const float iv = (float)gact[b * G4 + j];
      const float fv = (float)gact[b * G4 + 512 + j];
      const float gv = (float)gact[b * G4 + 1024 + j];
      const float ov = (float)gact[b * G4 + 1536 + j];
      c[q] = fv * c[q] + iv * gv;
      const float hn = ov * tanhf(c[q]);
      outs[id] = hn;                      // hidden_seq[s][b][j]
      hbf[id] = f2bf(hn);                 // feed next step's GEMM
      if (s == SSZ - 1) {
        out[(size_t)SSZ * BSZ * HSZ + id] = hn;               // h_last
        out[(size_t)SSZ * BSZ * HSZ + BSZ * HSZ + id] = c[q]; // c_last
      }
    }
    // loop back: top-of-loop tensorcnt wait + barrier covers all hazards
  }
}

// ---------------------------------------------------------------------------
// Host-side launcher. Workspace layout (bytes):
//   [0,        2MB)  Wx bf16 [2048][512]
//   [2MB,      4MB)  Wh bf16 [2048][512]
//   [4MB,      +8KB) fused bias f32 [2048]
//   [5MB,     69MB)  X bf16 [32][2048][512]
//   [69MB,   325MB)  x_proj bf16 [2048][32][2048]
// ---------------------------------------------------------------------------
extern "C" void kernel_launch(void* const* d_in, const int* in_sizes, int n_in,
                              void* d_out, int out_size, void* d_ws, size_t ws_size,
                              hipStream_t stream) {
  (void)in_sizes; (void)n_in; (void)out_size; (void)ws_size;
  const float* inputs = (const float*)d_in[0];
  const float* w_ii = (const float*)d_in[1];
  const float* w_hi = (const float*)d_in[2];
  const float* b_ii = (const float*)d_in[3];
  const float* b_hi = (const float*)d_in[4];
  const float* w_if = (const float*)d_in[5];
  const float* w_hf = (const float*)d_in[6];
  const float* b_if = (const float*)d_in[7];
  const float* b_hf = (const float*)d_in[8];
  const float* w_io = (const float*)d_in[9];
  const float* w_ho = (const float*)d_in[10];
  const float* b_io = (const float*)d_in[11];
  const float* b_ho = (const float*)d_in[12];
  const float* w_ig = (const float*)d_in[13];
  const float* w_hg = (const float*)d_in[14];
  const float* b_ig = (const float*)d_in[15];
  const float* b_hg = (const float*)d_in[16];

  char* ws = (char*)d_ws;
  unsigned short* Wx   = (unsigned short*)(ws);
  unsigned short* Wh   = (unsigned short*)(ws + (size_t)2 * 1024 * 1024);
  float*          bias = (float*)(ws + (size_t)4 * 1024 * 1024);
  unsigned short* Xb   = (unsigned short*)(ws + (size_t)5 * 1024 * 1024);
  unsigned short* xpro = (unsigned short*)(ws + (size_t)69 * 1024 * 1024);

  // Prep: pack weights (gate order i,f,g,o), fuse biases, convert inputs.
  lstm_pack_w<<<(G4 * ISZ) / 256, 256, 0, stream>>>(w_ii, w_if, w_ig, w_io, Wx);
  lstm_pack_w<<<(G4 * HSZ) / 256, 256, 0, stream>>>(w_hi, w_hf, w_hg, w_ho, Wh);
  lstm_pack_bias<<<G4 / 256, 256, 0, stream>>>(b_ii, b_hi, b_if, b_hf,
                                               b_ig, b_hg, b_io, b_ho, bias);
  lstm_conv_x<<<(BSZ * SSZ * ISZ) / 256, 256, 0, stream>>>(inputs, Xb, BSZ * SSZ * ISZ);

  // Big parallel input projection.
  dim3 grid(SSZ, 2, 2);
  lstm_xproj_gemm<<<grid, 256, 0, stream>>>(Xb, Wx, bias, xpro);

  // Sequential scan: one persistent workgroup, 288 KB dynamic LDS.
  size_t smem = (size_t)BSZ * G4 * 2 + (size_t)BSZ * HSZ * 2 + (size_t)BSZ * G4 * 2;
  lstm_scan<<<1, 1024, smem, stream>>>(Wh, xpro, (float*)d_out);
}
